// EmbeddingD_45621142618842
// MI455X (gfx1250) — matrix operand order repaired and checked by reading, not compiled, not run on hardware
//
#include <hip/hip_runtime.h>
#include <hip/hip_bf16.h>
#include <math.h>

// ---------------------------------------------------------------------------
// MGCN-style disease GCN pipeline for MI455X (gfx1250).
// GEMMs run on v_wmma_f32_16x16x32_bf16. A tiles: fp32 -> bf16 converted at
// staging time (cooperative, outside the WMMA dep chain). B (weights): one-time
// transpose+convert to bf16 [n][k] in global memory, then staged per-tile with
// gfx1250 async VMEM->LDS (global_load_async_to_lds_b128 / ASYNCcnt) -- no
// conversion, no VGPR round trip. Sparse GCN aggregation uses global f32
// atomics (targets are L2-resident: 16MB << 192MB L2).
// Workspace: 7 * N*FD fp32 + 6 * FD*FD bf16 + N*FD bf16 + 3*N f32  (~123 MB).
// ---------------------------------------------------------------------------

#define FDIM 512

typedef __attribute__((ext_vector_type(16))) __bf16 v16bf;
typedef __attribute__((ext_vector_type(8)))  float  v8f;

union FragAB { v16bf v; uint32_t u[8]; };
union FragC  { v8f  v; float f[8]; };

__device__ __forceinline__ uint16_t f2bf(float f) {
  uint32_t u = __builtin_bit_cast(uint32_t, f);
  uint32_t r = u + 0x7FFFu + ((u >> 16) & 1u);   // round-to-nearest-even
  return (uint16_t)(r >> 16);
}

// gfx1250 async global->LDS copy, 16B per lane. lds_off = LDS byte address
// (low 32 bits of a flat shared-aperture pointer), gaddr = global address.
__device__ __forceinline__ void async_ld_b128(uint32_t lds_off, const void* gaddr) {
  asm volatile("global_load_async_to_lds_b128 %0, %1, off"
               :: "v"(lds_off), "v"((unsigned long long)(uintptr_t)gaddr)
               : "memory");
}
__device__ __forceinline__ void wait_asynccnt0() {
  asm volatile("s_wait_asynccnt 0" ::: "memory");
}

// ---------------------------------------------------------------------------
// Tiled WMMA GEMM: C[M,Nc] = A[M,K] * B[K,Nc] (+bias[n]).  A fp32 row-major,
// B pre-converted bf16 transposed: Bt[n*K + k].  M,Nc multiples of 128,
// K multiple of 32 (true here: M=8192, Nc=512, K in {512, 8192}).
// Block: 256 threads (8 waves). Tile 128x128, K-chunk 32.
// Wave (wm in 0..3, wn in 0..1) computes rows [wm*32,+32) x cols [wn*64,+64)
// as 2x4 WMMA 16x16 tiles.
// ---------------------------------------------------------------------------
#define BM 128
#define BN 128
#define BK 32
#define LDA_S 40   // bf16 elements per A LDS row (32 + 8 pad, keeps 16B align)
#define LDB_S 40   // bf16 elements per B LDS row (32 + 8 pad, keeps 16B align)

__global__ __launch_bounds__(256)
void wmma_gemm_bf16(const float* __restrict__ A, const uint16_t* __restrict__ Bt,
                    float* __restrict__ C, const float* __restrict__ bias,
                    int M, int Nc, int K)
{
  __shared__ uint16_t As[BM * LDA_S];          // [m][k] bf16 (converted)
  __shared__ uint16_t Bs[BN * LDB_S];          // [n][k] bf16 (async-staged)

  const int tid   = threadIdx.x;
  const int m0    = blockIdx.x * BM;
  const int n0    = blockIdx.y * BN;
  const int wid   = tid >> 5;
  const int lane  = tid & 31;
  const int wm    = wid & 3;                   // 0..3 : M group
  const int wn    = wid >> 2;                  // 0..1 : N group
  const int lhalf = lane >> 4;                 // 0/1
  const int l15   = lane & 15;

  const uint32_t b_base = (uint32_t)(uintptr_t)&Bs[0];

  FragC acc[2][4];
  for (int mt = 0; mt < 2; ++mt)
    for (int nt = 0; nt < 4; ++nt)
      for (int r = 0; r < 8; ++r) acc[mt][nt].f[r] = 0.0f;

  for (int k0 = 0; k0 < K; k0 += BK) {
    __syncthreads();
    // ---- stage B tile [n0..n0+127][k0..k0+31] bf16 via async VMEM->LDS ----
    {
      const int rr  = tid >> 2;                // 0..63
      const int c16 = (tid & 3) * 8;           // 0,8,16,24 (bf16 elems, 16B)
      for (int p = 0; p < 2; ++p) {
        const int row = rr + p * 64;
        const uint16_t* gp = Bt + (size_t)(n0 + row) * K + k0 + c16;
        const uint32_t lo = b_base + (uint32_t)(row * LDB_S + c16) * 2u;
        async_ld_b128(lo, gp);
      }
    }
    // ---- stage A tile 128x32: fp32 -> bf16, overlapped with async B DMA ----
    {
      const int r0 = tid >> 3;                 // 0..31
      const int cw = (tid & 7) * 4;            // 0..28
      if (k0 + BK < K)                         // prefetch next A K-chunk
        __builtin_prefetch(A + (size_t)(m0 + r0) * K + k0 + BK + cw, 0, 1);
      for (int rr = r0; rr < BM; rr += 32) {
        const float4 v = *(const float4*)(A + (size_t)(m0 + rr) * K + k0 + cw);
        uint16_t* d = &As[rr * LDA_S + cw];
        d[0] = f2bf(v.x); d[1] = f2bf(v.y); d[2] = f2bf(v.z); d[3] = f2bf(v.w);
      }
    }
    wait_asynccnt0();                          // our async B loads landed
    __syncthreads();                           // everyone's tiles visible

    // ---- build fragments (CDNA5 16-bit A 16x32 / B 32x16 layouts) ----
    FragAB a[2], b[4];
    const int kbA = lhalf * 8;                 // lanes 16-31 start at K=8
    for (int mt = 0; mt < 2; ++mt) {
      const int m = wm * 32 + mt * 16 + l15;
      const uint32_t* p = (const uint32_t*)&As[m * LDA_S + kbA];
      *(uint4*)&a[mt].u[0] = *(const uint4*)(p);       // K = kb .. kb+7
      *(uint4*)&a[mt].u[4] = *(const uint4*)(p + 8);   // K = kb+16 .. kb+23
    }
    const int kbB = lhalf * 8;                 // lanes 16-31 start at K=8
    for (int nt = 0; nt < 4; ++nt) {
      const int nn = wn * 64 + nt * 16 + l15;
      const uint32_t* p = (const uint32_t*)&Bs[nn * LDB_S + kbB];
      *(uint4*)&b[nt].u[0] = *(const uint4*)(p);       // halves 0..7
      *(uint4*)&b[nt].u[4] = *(const uint4*)(p + 8);   // K = kb+16 .. kb+23
    }

    for (int mt = 0; mt < 2; ++mt)
      for (int nt = 0; nt < 4; ++nt)
        acc[mt][nt].v = __builtin_amdgcn_wmma_f32_16x16x32_bf16(
            false, a[mt].v, false, b[nt].v, (short)0, acc[mt][nt].v,
            false, false);
  }

  // ---- epilogue: C/D layout -> global ----
  for (int mt = 0; mt < 2; ++mt) {
    const int mb = m0 + wm * 32 + mt * 16 + lhalf * 8;
    for (int nt = 0; nt < 4; ++nt) {
      const int nn = n0 + wn * 64 + nt * 16 + l15;
      const float bv = bias ? bias[nn] : 0.0f;
      for (int r = 0; r < 8; ++r)
        C[(size_t)(mb + r) * Nc + nn] = acc[mt][nt].f[r] + bv;
    }
  }
}

// One-time weight prep: Wt[n*K + k] = bf16(W[k*Nc + n])   (W is [K,Nc] fp32)
__global__ void convT_bf16(const float* __restrict__ W, uint16_t* __restrict__ Wt,
                           int K, int Nc)
{
  long long i = (long long)blockIdx.x * blockDim.x + threadIdx.x;
  if (i >= (long long)K * Nc) return;
  int k = (int)(i / Nc);
  int n = (int)(i % Nc);
  Wt[(size_t)n * K + k] = f2bf(W[i]);
}

// ---------------------------------------------------------------------------
// Small helper kernels
// ---------------------------------------------------------------------------
__global__ void fill_f32(float* __restrict__ p, float v, int n) {
  int i = blockIdx.x * blockDim.x + threadIdx.x;
  if (i < n) p[i] = v;
}

__global__ void deg_accum(float* __restrict__ deg, const int* __restrict__ edges,
                          const float* __restrict__ w, int E) {
  int e = blockIdx.x * blockDim.x + threadIdx.x;
  if (e >= E) return;
  int d = edges[E + e];                        // edges[1][e] = dst
  float we = w ? w[e] : 1.0f;
  atomicAdd(&deg[d], we);
}

__global__ void deg_to_dinv(float* __restrict__ deg, int n) {
  int i = blockIdx.x * blockDim.x + threadIdx.x;
  if (i >= n) return;
  float d = deg[i];
  deg[i] = (d > 0.0f) ? rsqrtf(fmaxf(d, 1e-12f)) : 0.0f;
}

// Edge aggregation: acc[dst] += norm_e * h[src]. 128 threads per edge, 4 f32
// per thread -> global_atomic_add_f32 scatter (L2-resident targets).
__global__ __launch_bounds__(256)
void agg_edges(float* __restrict__ acc, const float* __restrict__ h,
               const int* __restrict__ edges, const float* __restrict__ w,
               const float* __restrict__ dinv, int E)
{
  long long t = (long long)blockIdx.x * blockDim.x + threadIdx.x;
  int e = (int)(t >> 7);
  if (e >= E) return;
  int f = ((int)t & 127) * 4;
  int s = edges[e];
  int d = edges[E + e];
  float we = w ? w[e] : 1.0f;
  float norm = dinv[s] * we * dinv[d];
  const float4 hv = *(const float4*)(h + (size_t)s * FDIM + f);
  float* dst = acc + (size_t)d * FDIM + f;
  atomicAdd(dst + 0, norm * hv.x);
  atomicAdd(dst + 1, norm * hv.y);
  atomicAdd(dst + 2, norm * hv.z);
  atomicAdd(dst + 3, norm * hv.w);
}

// out = relu(acc + dinv[n]^2 * h + bias[f])  (+ residual)   [in-place ok]
__global__ void gcn_finalize(float* __restrict__ out, const float* __restrict__ acc,
                             const float* __restrict__ h, const float* __restrict__ dinv,
                             const float* __restrict__ bias,
                             const float* __restrict__ residual, int n)
{
  int i = blockIdx.x * blockDim.x + threadIdx.x;
  if (i >= n) return;
  int node = i / FDIM;
  int f    = i % FDIM;
  float dv = dinv[node];
  float v = acc[i] + dv * dv * h[i] + bias[f];
  v = fmaxf(v, 0.0f);
  if (residual) v += residual[i];
  out[i] = v;
}

// Cross-stitch: compose the 4 fixed 2x2 units into one matrix, apply elementwise.
__global__ void cross_stitch(float* __restrict__ yg, float* __restrict__ mp,
                             const float* __restrict__ cs, int n)
{
  float m00 = 1.f, m01 = 0.f, m10 = 0.f, m11 = 0.f;
  for (int i = 0; i < 4; ++i) {
    float c00 = cs[i * 4 + 0], c01 = cs[i * 4 + 1];
    float c10 = cs[i * 4 + 2], c11 = cs[i * 4 + 3];
    float n00 = c00 * m00 + c01 * m10;
    float n01 = c00 * m01 + c01 * m11;
    float n10 = c10 * m00 + c11 * m10;
    float n11 = c10 * m01 + c11 * m11;
    m00 = n00; m01 = n01; m10 = n10; m11 = n11;
  }
  int i = blockIdx.x * blockDim.x + threadIdx.x;
  if (i >= n) return;
  float a = yg[i], b = mp[i];
  yg[i] = m00 * a + m01 * b;
  mp[i] = m10 * a + m11 * b;
}

__global__ __launch_bounds__(256)
void pool_sum(const float* __restrict__ x0, const float* __restrict__ x1,
              const float* __restrict__ x2, const float* __restrict__ x3,
              float* __restrict__ pooled, int n)
{
  const float* xs[4] = {x0, x1, x2, x3};
  const float* x = xs[blockIdx.y];
  __shared__ float red[256];
  float s = 0.0f;
  for (int i = blockIdx.x * blockDim.x + threadIdx.x; i < n;
       i += gridDim.x * blockDim.x) s += x[i];
  red[threadIdx.x] = s;
  __syncthreads();
  for (int off = 128; off > 0; off >>= 1) {
    if (threadIdx.x < off) red[threadIdx.x] += red[threadIdx.x + off];
    __syncthreads();
  }
  if (threadIdx.x == 0) atomicAdd(&pooled[blockIdx.y], red[0]);
}

// Tiny MLP 4 -> 20 -> 4 with sigmoid: single-thread is fine.
__global__ void attention(const float* __restrict__ pooled,
                          const float* __restrict__ fc1W, const float* __restrict__ fc1b,
                          const float* __restrict__ fc2W, const float* __restrict__ fc2b,
                          float* __restrict__ att, float invn)
{
  if (threadIdx.x != 0 || blockIdx.x != 0) return;
  float p[4];
  for (int c = 0; c < 4; ++c) p[c] = pooled[c] * invn;
  float hbuf[20];
  for (int j = 0; j < 20; ++j) {
    float s = fc1b[j];
    for (int c = 0; c < 4; ++c) s += p[c] * fc1W[c * 20 + j];
    hbuf[j] = fmaxf(s, 0.0f);
  }
  for (int c = 0; c < 4; ++c) {
    float s = fc2b[c];
    for (int j = 0; j < 20; ++j) s += hbuf[j] * fc2W[j * 4 + c];
    att[c] = 1.0f / (1.0f + expf(-s));
  }
}

// Fused attention-scale + relu + 1x1 conv (4ch -> 1ch). Output layout == y.T.
__global__ void combine(const float* __restrict__ x0, const float* __restrict__ x1,
                        const float* __restrict__ x2, const float* __restrict__ x3,
                        const float* __restrict__ att, const float* __restrict__ cnnw,
                        const float* __restrict__ cnnb, float* __restrict__ out, int n)
{
  int i = blockIdx.x * blockDim.x + threadIdx.x;
  if (i >= n) return;
  float v = cnnb[0];
  v += cnnw[0] * fmaxf(att[0] * x0[i], 0.0f);
  v += cnnw[1] * fmaxf(att[1] * x1[i], 0.0f);
  v += cnnw[2] * fmaxf(att[2] * x2[i], 0.0f);
  v += cnnw[3] * fmaxf(att[3] * x3[i], 0.0f);
  out[i] = v;
}

// ---------------------------------------------------------------------------
extern "C" void kernel_launch(void* const* d_in, const int* in_sizes, int n_in,
                              void* d_out, int out_size, void* d_ws, size_t ws_size,
                              hipStream_t stream)
{
  (void)n_in; (void)out_size; (void)ws_size;
  const float* x_d     = (const float*)d_in[0];
  const int*   edges_s = (const int*)  d_in[1];
  const float* w_s     = (const float*)d_in[2];
  const int*   edges_g = (const int*)  d_in[3];
  const float* w_g     = (const float*)d_in[4];
  const int*   edges_I = (const int*)  d_in[5];
  const float* reach   = (const float*)d_in[6];
  const float* W1s = (const float*)d_in[7],  *b1s = (const float*)d_in[8];
  const float* W2s = (const float*)d_in[9],  *b2s = (const float*)d_in[10];
  const float* W1g = (const float*)d_in[11], *b1g = (const float*)d_in[12];
  const float* W2g = (const float*)d_in[13], *b2g = (const float*)d_in[14];
  const float* WI1 = (const float*)d_in[15], *bI1 = (const float*)d_in[16];
  const float* WI2 = (const float*)d_in[17], *bI2 = (const float*)d_in[18];
  const float* fcW  = (const float*)d_in[19];
  const float* fcb  = (const float*)d_in[20];
  const float* cs   = (const float*)d_in[21];
  const float* fc1W = (const float*)d_in[22];
  const float* fc1b = (const float*)d_in[23];
  const float* fc2W = (const float*)d_in[24];
  const float* fc2b = (const float*)d_in[25];
  const float* cnnw = (const float*)d_in[26];
  const float* cnnb = (const float*)d_in[27];

  const int NN = in_sizes[0] / FDIM;        // 8192
  const int E  = in_sizes[2];               // 262144
  const int NFD = NN * FDIM;                // 4,194,304
  const size_t NFDb = (size_t)NFD * sizeof(float);
  const size_t WTb  = (size_t)FDIM * FDIM * sizeof(uint16_t);  // 512KB

  // workspace carve
  char* w = (char*)d_ws;
  float* bufH = (float*)w; w += NFDb;
  float* y_s1 = (float*)w; w += NFDb;
  float* y_g1 = (float*)w; w += NFDb;
  float* mp1  = (float*)w; w += NFDb;
  float* y_s2 = (float*)w; w += NFDb;
  float* y_g2 = (float*)w; w += NFDb;
  float* feat = (float*)w; w += NFDb;
  uint16_t* Wt1s = (uint16_t*)w; w += WTb;
  uint16_t* Wt2s = (uint16_t*)w; w += WTb;
  uint16_t* Wt1g = (uint16_t*)w; w += WTb;
  uint16_t* Wt2g = (uint16_t*)w; w += WTb;
  uint16_t* WtI1 = (uint16_t*)w; w += WTb;
  uint16_t* WtI2 = (uint16_t*)w; w += WTb;
  uint16_t* fcWt = (uint16_t*)w; w += (size_t)NFD * sizeof(uint16_t);  // [FD][N]
  float* dinv_s = (float*)w; w += (size_t)NN * sizeof(float);
  float* dinv_g = (float*)w; w += (size_t)NN * sizeof(float);
  float* dinv_I = (float*)w; w += (size_t)NN * sizeof(float);
  float* pooled = (float*)w;                 // 4 floats
  float* att    = pooled + 4;                // 4 floats

  float* y_out   = (float*)d_out;            // y.T  [N, FD]
  float* mp2_out = (float*)d_out + NFD;      // mp2  [N, FD]

  const dim3 blk(256);
  const dim3 gElem((NFD + 255) / 256);
  const dim3 gNode((NN + 255) / 256);
  const dim3 gEdge((E + 255) / 256);
  const dim3 gAgg((unsigned)(((long long)E * 128 + 255) / 256));
  const dim3 gGemm(NN / BM, FDIM / BN);
  const dim3 gWsm((FDIM * FDIM + 255) / 256);

  // ---- one-time weight transpose+convert to bf16 [n][k] ----
  convT_bf16<<<gWsm, blk, 0, stream>>>(W1s, Wt1s, FDIM, FDIM);
  convT_bf16<<<gWsm, blk, 0, stream>>>(W2s, Wt2s, FDIM, FDIM);
  convT_bf16<<<gWsm, blk, 0, stream>>>(W1g, Wt1g, FDIM, FDIM);
  convT_bf16<<<gWsm, blk, 0, stream>>>(W2g, Wt2g, FDIM, FDIM);
  convT_bf16<<<gWsm, blk, 0, stream>>>(WI1, WtI1, FDIM, FDIM);
  convT_bf16<<<gWsm, blk, 0, stream>>>(WI2, WtI2, FDIM, FDIM);
  convT_bf16<<<gElem, blk, 0, stream>>>(fcW, fcWt, NN, FDIM);

  // ---- symmetric-normalization degree factors (self-loop weight 1) ----
  fill_f32<<<gNode, blk, 0, stream>>>(dinv_s, 1.0f, NN);
  fill_f32<<<gNode, blk, 0, stream>>>(dinv_g, 1.0f, NN);
  fill_f32<<<gNode, blk, 0, stream>>>(dinv_I, 1.0f, NN);
  deg_accum<<<gEdge, blk, 0, stream>>>(dinv_s, edges_s, w_s, E);
  deg_accum<<<gEdge, blk, 0, stream>>>(dinv_g, edges_g, w_g, E);
  deg_accum<<<gEdge, blk, 0, stream>>>(dinv_I, edges_I, nullptr, E);
  deg_to_dinv<<<gNode, blk, 0, stream>>>(dinv_s, NN);
  deg_to_dinv<<<gNode, blk, 0, stream>>>(dinv_g, NN);
  deg_to_dinv<<<gNode, blk, 0, stream>>>(dinv_I, NN);

  // ---- y_s1 = relu(gcn(x, W1s)) ----
  wmma_gemm_bf16<<<gGemm, blk, 0, stream>>>(x_d, Wt1s, bufH, nullptr, NN, FDIM, FDIM);
  hipMemsetAsync(y_s1, 0, NFDb, stream);
  agg_edges<<<gAgg, blk, 0, stream>>>(y_s1, bufH, edges_s, w_s, dinv_s, E);
  gcn_finalize<<<gElem, blk, 0, stream>>>(y_s1, y_s1, bufH, dinv_s, b1s, nullptr, NFD);

  // ---- y_g1 = relu(gcn(x, W1g)) ----
  wmma_gemm_bf16<<<gGemm, blk, 0, stream>>>(x_d, Wt1g, bufH, nullptr, NN, FDIM, FDIM);
  hipMemsetAsync(y_g1, 0, NFDb, stream);
  agg_edges<<<gAgg, blk, 0, stream>>>(y_g1, bufH, edges_g, w_g, dinv_g, E);
  gcn_finalize<<<gElem, blk, 0, stream>>>(y_g1, y_g1, bufH, dinv_g, b1g, nullptr, NFD);

  // ---- feat = reach @ fcW + fcb   (the big 8192x8192x512 GEMM) ----
  wmma_gemm_bf16<<<gGemm, blk, 0, stream>>>(reach, fcWt, feat, fcb, NN, FDIM, NN);

  // ---- mp1 = relu(gcn(feat, WI1)) ----
  wmma_gemm_bf16<<<gGemm, blk, 0, stream>>>(feat, WtI1, bufH, nullptr, NN, FDIM, FDIM);
  hipMemsetAsync(mp1, 0, NFDb, stream);
  agg_edges<<<gAgg, blk, 0, stream>>>(mp1, bufH, edges_I, nullptr, dinv_I, E);
  gcn_finalize<<<gElem, blk, 0, stream>>>(mp1, mp1, bufH, dinv_I, bI1, nullptr, NFD);

  // ---- cross-stitch fusion (4 units composed) ----
  cross_stitch<<<gElem, blk, 0, stream>>>(y_g1, mp1, cs, NFD);

  // ---- y_s2 = relu(gcn(y_s1, W2s)) ----
  wmma_gemm_bf16<<<gGemm, blk, 0, stream>>>(y_s1, Wt2s, bufH, nullptr, NN, FDIM, FDIM);
  hipMemsetAsync(y_s2, 0, NFDb, stream);
  agg_edges<<<gAgg, blk, 0, stream>>>(y_s2, bufH, edges_s, w_s, dinv_s, E);
  gcn_finalize<<<gElem, blk, 0, stream>>>(y_s2, y_s2, bufH, dinv_s, b2s, nullptr, NFD);

  // ---- mp2 = relu(gcn(mp1, WI2)) + mp1  -> second half of d_out ----
  wmma_gemm_bf16<<<gGemm, blk, 0, stream>>>(mp1, WtI2, bufH, nullptr, NN, FDIM, FDIM);
  hipMemsetAsync(mp2_out, 0, NFDb, stream);
  agg_edges<<<gAgg, blk, 0, stream>>>(mp2_out, bufH, edges_I, nullptr, dinv_I, E);
  gcn_finalize<<<gElem, blk, 0, stream>>>(mp2_out, mp2_out, bufH, dinv_I, bI2, mp1, NFD);

  // ---- y_g2 = relu(gcn(y_g1_stitched, W2g)) ----
  wmma_gemm_bf16<<<gGemm, blk, 0, stream>>>(y_g1, Wt2g, bufH, nullptr, NN, FDIM, FDIM);
  hipMemsetAsync(y_g2, 0, NFDb, stream);
  agg_edges<<<gAgg, blk, 0, stream>>>(y_g2, bufH, edges_g, w_g, dinv_g, E);
  gcn_finalize<<<gElem, blk, 0, stream>>>(y_g2, y_g2, bufH, dinv_g, b2g, nullptr, NFD);

  // ---- channel attention + fused 1x1 conv combine ----
  hipMemsetAsync(pooled, 0, 8 * sizeof(float), stream);
  pool_sum<<<dim3(256, 4), blk, 0, stream>>>(y_s1, y_s2, y_g1, y_g2, pooled, NFD);
  attention<<<1, 32, 0, stream>>>(pooled, fc1W, fc1b, fc2W, fc2b, att,
                                  1.0f / (float)NFD);
  combine<<<gElem, blk, 0, stream>>>(y_s1, y_s2, y_g1, y_g2, att, cnnw, cnnb,
                                     y_out, NFD);
}
